// MambaSSM_51187420234048
// MI455X (gfx1250) — matrix operand order repaired and checked by reading, not compiled, not run on hardware
//
#include <hip/hip_runtime.h>
#include <hip/hip_bf16.h>
#include <math.h>

// ---------------------------------------------------------------------------
// MambaSSM on gfx1250: f16 WMMA GEMMs (fp32 accum) + chunked parallel scan
// with global_prefetch_b8 read-ahead on the streaming scan passes.
// ---------------------------------------------------------------------------

typedef __attribute__((ext_vector_type(16))) _Float16 v16h;
typedef __attribute__((ext_vector_type(8)))  float    v8f;

#define TT      65536      // sequence length
#define OBS     64
#define EMB     8
#define INDIM   72
#define KP0     96         // INDIM padded to multiple of 32
#define HID     256
#define NOBJ    4
#define MT      (TT/16)    // 4096 M-tiles
#define NT      (HID/16)   // 16  N-tiles
#define NCHUNK  256
#define CHUNK   (TT/NCHUNK) // 256
#define PF      16         // scan prefetch distance (iterations ahead)

// ---------------------------------------------------------------------------
// A-fragment packing (16x32 f16 tile, ISA 7.12.2):
//   lanes 0-15 : M = lane,   K in {0..7} (halves 0-7) and {16..23} (halves 8-15)
//   lanes 16-31: M = lane-16,K in {8..15}             and {24..31}
// Packed address: ((mtile*KT + kt)*32 + lane) * 16 halves  (one v16h per lane)
// ---------------------------------------------------------------------------

// Build + pack layer-0 input: X = [obs | emb[action] | zero-pad] -> f16 frags
__global__ __launch_bounds__(256)
void pack_x0_kernel(const float* __restrict__ obs, const int* __restrict__ act,
                    const float* __restrict__ emb, v16h* __restrict__ dst)
{
    const int KT = KP0 / 32; // 3
    int gid  = blockIdx.x * blockDim.x + threadIdx.x;
    int lane = gid & 31;
    int tile = gid >> 5;
    if (tile >= MT * KT) return;
    int kt    = tile % KT;
    int mtile = tile / KT;
    int row   = mtile * 16 + (lane & 15);
    int kbase = kt * 32 + ((lane < 16) ? 0 : 8);
    int a     = act[row];
    v16h frag;
#pragma unroll
    for (int e = 0; e < 16; ++e) {
        int k = kbase + ((e < 8) ? e : (e + 8));
        float v;
        if (k < OBS)        v = obs[(size_t)row * OBS + k];
        else if (k < INDIM) v = emb[a * EMB + (k - OBS)];
        else                v = 0.0f;
        frag[e] = (_Float16)v;
    }
    dst[(size_t)tile * 32 + lane] = frag;
}

// Pack a dense fp32 [T, Ksrc] activation matrix into A-fragments (Ksrc = KT*32)
__global__ __launch_bounds__(256)
void pack_dense_kernel(const float* __restrict__ src, v16h* __restrict__ dst,
                       int KT, int Ksrc)
{
    int gid  = blockIdx.x * blockDim.x + threadIdx.x;
    int lane = gid & 31;
    int tile = gid >> 5;
    if (tile >= MT * KT) return;
    int kt    = tile % KT;
    int mtile = tile / KT;
    int row   = mtile * 16 + (lane & 15);
    int kbase = kt * 32 + ((lane < 16) ? 0 : 8);
    v16h frag;
#pragma unroll
    for (int e = 0; e < 16; ++e) {
        int k = kbase + ((e < 8) ? e : (e + 8));
        frag[e] = (_Float16)((k < Ksrc) ? src[(size_t)row * Ksrc + k] : 0.0f);
    }
    dst[(size_t)tile * 32 + lane] = frag;
}

// ---------------------------------------------------------------------------
// B-fragment packing (32x16 f16 = K x N, ISA 7.12.2 / sparse-B table):
//   lanes 0-15 : N = lane,    halves 0-7 = K 0..7,  halves 8-15 = K 16..23
//   lanes 16-31: N = lane-16, halves 0-7 = K 8..15, halves 8-15 = K 24..31
// B[k][n] = W[n][k]  (we compute Y = X * W^T).
// Packed address: (((m*NT + nt)*KT + kt)*32 + lane) * 16 halves
// ---------------------------------------------------------------------------
__global__ __launch_bounds__(256)
void pack_w_kernel(const float* __restrict__ W0, const float* __restrict__ W1,
                   const float* __restrict__ W2, const float* __restrict__ W3,
                   int Kin, int KT, v16h* __restrict__ dst)
{
    int gid  = blockIdx.x * blockDim.x + threadIdx.x;
    int lane = gid & 31;
    int tile = gid >> 5;
    if (tile >= 4 * NT * KT) return;
    int kt = tile % KT;
    int nt = (tile / KT) % NT;
    int m  = tile / (KT * NT);
    const float* W = (m == 0) ? W0 : (m == 1) ? W1 : (m == 2) ? W2 : W3;
    int n      = nt * 16 + (lane & 15);
    int kstart = kt * 32 + ((lane < 16) ? 0 : 8);
    v16h frag;
#pragma unroll
    for (int e = 0; e < 16; ++e) {
        int k = kstart + ((e < 8) ? e : (e + 8));
        frag[e] = (_Float16)((k < Kin) ? W[(size_t)n * Kin + k] : 0.0f);
    }
    dst[(size_t)tile * 32 + lane] = frag;
}

// ---------------------------------------------------------------------------
// Fused 4-matrix WMMA GEMM + SSM gate epilogue.
// Per wave: one 16-row M-tile; A K-fragments held in registers across the
// 16 N-tiles; 4 accumulators (x_state, B, C, delta) share each A fragment.
// Epilogue: Abar = exp(sigmoid(delta) * -exp(Alog[h])), Bx = B*x_state, C.
// ---------------------------------------------------------------------------
template <int KT>
__global__ __launch_bounds__(256)
void ssm_gemm_wmma(const v16h* __restrict__ Xp, const v16h* __restrict__ Wp,
                   const float* __restrict__ Alog,
                   float* __restrict__ Abar, float* __restrict__ Bx,
                   float* __restrict__ Cc)
{
    const int lane  = threadIdx.x & 31;
    const int wave  = threadIdx.x >> 5;
    const int mtile = blockIdx.x * 8 + wave;

    v16h a[KT];
    const v16h* abase = Xp + (size_t)mtile * KT * 32 + lane;
#pragma unroll
    for (int kt = 0; kt < KT; ++kt) a[kt] = abase[kt * 32];

    const int colN   = lane & 15;
    const int rowOff = (lane < 16) ? 0 : 8;
    const int tbase  = mtile * 16 + rowOff;
    const size_t mstride = (size_t)NT * KT * 32;  // v16h stride between matrices

    for (int nt = 0; nt < NT; ++nt) {
        v8f accX = {}, accB = {}, accC = {}, accD = {};
        const v16h* wbase = Wp + (size_t)nt * KT * 32 + lane;
#pragma unroll
        for (int kt = 0; kt < KT; ++kt) {
            v16h b0 = wbase[0 * mstride + kt * 32];
            v16h b1 = wbase[1 * mstride + kt * 32];
            v16h b2 = wbase[2 * mstride + kt * 32];
            v16h b3 = wbase[3 * mstride + kt * 32];
            accX = __builtin_amdgcn_wmma_f32_16x16x32_f16(false, a[kt], false, b0,
                                                          (short)0, accX, false, false);
            accB = __builtin_amdgcn_wmma_f32_16x16x32_f16(false, a[kt], false, b1,
                                                          (short)0, accB, false, false);
            accC = __builtin_amdgcn_wmma_f32_16x16x32_f16(false, a[kt], false, b2,
                                                          (short)0, accC, false, false);
            accD = __builtin_amdgcn_wmma_f32_16x16x32_f16(false, a[kt], false, b3,
                                                          (short)0, accD, false, false);
        }
        int   h  = nt * 16 + colN;
        float Ah = -expf(Alog[h]);
#pragma unroll
        for (int v = 0; v < 8; ++v) {
            size_t idx   = (size_t)(tbase + v) * HID + h;
            float  delta = 1.0f / (1.0f + expf(-accD[v]));
            Abar[idx] = expf(delta * Ah);
            Bx[idx]   = accB[v] * accX[v];
            Cc[idx]   = accC[v];
        }
    }
}

// ---------------------------------------------------------------------------
// Chunked parallel scan over h_t = a_t*h_{t-1} + bx_t  (per-channel).
// Each wave's 32 lanes cover a contiguous 128B slab per t-step; one-byte
// prefetches (global_prefetch_b8) pull the slab PF iterations ahead so the
// serial FMA chain never waits on HBM.
// ---------------------------------------------------------------------------
__global__ __launch_bounds__(HID)
void scan_pass1(const float* __restrict__ Abar, const float* __restrict__ Bx,
                float* __restrict__ ca, float* __restrict__ cb)
{
    int h = threadIdx.x, c = blockIdx.x;
    float pa = 1.0f, hb = 0.0f;
    size_t base = (size_t)c * CHUNK * HID + h;
#pragma unroll 4
    for (int t = 0; t < CHUNK; ++t) {
        if (t + PF < CHUNK) {
            __builtin_prefetch(&Abar[base + (size_t)(t + PF) * HID], 0, 3);
            __builtin_prefetch(&Bx[base + (size_t)(t + PF) * HID], 0, 3);
        }
        float av = Abar[base + (size_t)t * HID];
        float bv = Bx[base + (size_t)t * HID];
        pa *= av;
        hb = av * hb + bv;
    }
    ca[c * HID + h] = pa;
    cb[c * HID + h] = hb;
}

__global__ __launch_bounds__(HID)
void scan_pass2(const float* __restrict__ ca, const float* __restrict__ cb,
                float* __restrict__ pref)
{
    int h = threadIdx.x;
    float hin = 0.0f;
    for (int c = 0; c < NCHUNK; ++c) {
        pref[c * HID + h] = hin;
        hin = ca[c * HID + h] * hin + cb[c * HID + h];
    }
}

// Re-scan each chunk from its prefix and write y = C*h (Y may alias Cc).
__global__ __launch_bounds__(HID)
void scan_pass3(const float* __restrict__ Abar, const float* __restrict__ Bx,
                const float* Cc, const float* __restrict__ pref, float* Y)
{
    int h = threadIdx.x, c = blockIdx.x;
    float hs = pref[c * HID + h];
    size_t base = (size_t)c * CHUNK * HID + h;
#pragma unroll 4
    for (int t = 0; t < CHUNK; ++t) {
        size_t idx = base + (size_t)t * HID;
        if (t + PF < CHUNK) {
            __builtin_prefetch(&Abar[base + (size_t)(t + PF) * HID], 0, 3);
            __builtin_prefetch(&Bx[base + (size_t)(t + PF) * HID], 0, 3);
            __builtin_prefetch(&Cc[base + (size_t)(t + PF) * HID], 0, 3);
        }
        float av = Abar[idx], bv = Bx[idx];
        hs = av * hs + bv;
        Y[idx] = Cc[idx] * hs;
    }
}

// ---------------------------------------------------------------------------
// Output head: one wave per timestep, 4-way dot over H with shuffle reduce,
// then softplus(out)+1.
// ---------------------------------------------------------------------------
__global__ __launch_bounds__(256)
void head_kernel(const float* __restrict__ Y, const float* __restrict__ Wout,
                 const float* __restrict__ bout, float* __restrict__ out)
{
    int lane = threadIdx.x & 31;
    int wave = threadIdx.x >> 5;
    int t    = blockIdx.x * 8 + wave;
    const float* y = Y + (size_t)t * HID;
    __builtin_prefetch(&y[lane * 8], 0, 3);   // cover the 1KB row ahead of use
    float s0 = 0.f, s1 = 0.f, s2 = 0.f, s3 = 0.f;
    for (int j = lane; j < HID; j += 32) {
        float v = y[j];
        s0 += v * Wout[0 * HID + j];
        s1 += v * Wout[1 * HID + j];
        s2 += v * Wout[2 * HID + j];
        s3 += v * Wout[3 * HID + j];
    }
#pragma unroll
    for (int o = 16; o >= 1; o >>= 1) {
        s0 += __shfl_xor(s0, o, 32);
        s1 += __shfl_xor(s1, o, 32);
        s2 += __shfl_xor(s2, o, 32);
        s3 += __shfl_xor(s3, o, 32);
    }
    if (lane == 0) {
        float v[4] = {s0, s1, s2, s3};
#pragma unroll
        for (int o = 0; o < NOBJ; ++o) {
            float z  = v[o] + bout[o];
            float sp = (z > 20.0f) ? z : log1pf(expf(z));
            out[(size_t)t * NOBJ + o] = sp + 1.0f;
        }
    }
}

// ---------------------------------------------------------------------------
extern "C" void kernel_launch(void* const* d_in, const int* in_sizes, int n_in,
                              void* d_out, int out_size, void* d_ws, size_t ws_size,
                              hipStream_t stream) {
    (void)in_sizes; (void)n_in; (void)out_size; (void)ws_size;
    const float* obs  = (const float*)d_in[0];
    const int*   act  = (const int*)d_in[1];
    const float* emb  = (const float*)d_in[2];
    const float* Win0 = (const float*)d_in[3];
    const float* WB0  = (const float*)d_in[4];
    const float* WC0  = (const float*)d_in[5];
    const float* Wd0  = (const float*)d_in[6];
    const float* Alog0= (const float*)d_in[7];
    const float* Win1 = (const float*)d_in[8];
    const float* WB1  = (const float*)d_in[9];
    const float* WC1  = (const float*)d_in[10];
    const float* Wd1  = (const float*)d_in[11];
    const float* Alog1= (const float*)d_in[12];
    const float* Wout = (const float*)d_in[13];
    const float* bout = (const float*)d_in[14];
    float* out = (float*)d_out;

    // --- workspace carve-out (aligned to 256B) ---
    char*  ws = (char*)d_ws;
    size_t off = 0;
    auto carve = [&](size_t bytes) -> char* {
        char* p = ws + off;
        off = (off + bytes + 255) & ~(size_t)255;
        return p;
    };
    v16h*  Xp   = (v16h*)carve((size_t)TT * HID * 2);        // packed f16 activations (both layers)
    v16h*  Wp   = (v16h*)carve((size_t)4 * HID * HID * 2);   // packed f16 weights (re-packed per layer)
    float* Abar = (float*)carve((size_t)TT * HID * 4);
    float* Bx   = (float*)carve((size_t)TT * HID * 4);
    float* Cc   = (float*)carve((size_t)TT * HID * 4);       // C, then y in place
    float* ca   = (float*)carve((size_t)NCHUNK * HID * 4);
    float* cb   = (float*)carve((size_t)NCHUNK * HID * 4);
    float* pref = (float*)carve((size_t)NCHUNK * HID * 4);

    // ---------------- layer 0 (K padded 72 -> 96, KT = 3) ----------------
    {
        const int KT = 3;
        pack_x0_kernel<<<(MT * KT * 32) / 256, 256, 0, stream>>>(obs, act, emb, Xp);
        pack_w_kernel<<<(4 * NT * KT * 32 + 255) / 256, 256, 0, stream>>>(
            Win0, WB0, WC0, Wd0, INDIM, KT, Wp);
        ssm_gemm_wmma<3><<<MT / 8, 256, 0, stream>>>(Xp, Wp, Alog0, Abar, Bx, Cc);
        scan_pass1<<<NCHUNK, HID, 0, stream>>>(Abar, Bx, ca, cb);
        scan_pass2<<<1, HID, 0, stream>>>(ca, cb, pref);
        scan_pass3<<<NCHUNK, HID, 0, stream>>>(Abar, Bx, Cc, pref, Cc); // y0 -> Cc
    }

    // ---------------- layer 1 (K = 256, KT = 8) ----------------
    {
        const int KT = 8;
        pack_dense_kernel<<<(MT * KT * 32) / 256, 256, 0, stream>>>(Cc, Xp, KT, HID);
        pack_w_kernel<<<(4 * NT * KT * 32 + 255) / 256, 256, 0, stream>>>(
            Win1, WB1, WC1, Wd1, HID, KT, Wp);
        ssm_gemm_wmma<8><<<MT / 8, 256, 0, stream>>>(Xp, Wp, Alog1, Abar, Bx, Cc);
        scan_pass1<<<NCHUNK, HID, 0, stream>>>(Abar, Bx, ca, cb);
        scan_pass2<<<1, HID, 0, stream>>>(ca, cb, pref);
        scan_pass3<<<NCHUNK, HID, 0, stream>>>(Abar, Bx, Cc, pref, Cc); // y1 -> Cc
    }

    // ---------------- output head ----------------
    head_kernel<<<TT / 8, 256, 0, stream>>>(Cc, Wout, bout, out);
}
